// CrossAttention_19473381720410
// MI455X (gfx1250) — matrix operand order repaired and checked by reading, not compiled, not run on hardware
//
#include <hip/hip_runtime.h>

typedef __attribute__((ext_vector_type(16))) _Float16 v16h;
typedef __attribute__((ext_vector_type(8)))  _Float16 v8h;
typedef __attribute__((ext_vector_type(4)))  _Float16 v4h;
typedef __attribute__((ext_vector_type(2)))  _Float16 v2h;
typedef __attribute__((ext_vector_type(8)))  float    v8f;
typedef __attribute__((ext_vector_type(4)))  float    v4f;

namespace {
constexpr int Bc = 2;
constexpr int Tc = 2048;
constexpr int Hc = 16;
constexpr int Dc = 64;
constexpr int BM = 64;                  // query rows per workgroup (16 per wave)
constexpr int BN = 64;                  // kv rows per tile
constexpr int NWAVES   = BM / 16;       // 4
constexpr int NTHREADS = NWAVES * 32;   // 128
constexpr int QSTRIDE  = Hc * Dc;       // 1024 floats per t in q/out
constexpr int KVSTRIDE = 2 * Hc * Dc;   // 2048 floats per t in kv
constexpr int ROWH     = Dc + 8;        // 72 halves per LDS row (bank padding)
}

__device__ __forceinline__ v16h cat8(v8h lo, v8h hi) {
  return __builtin_shufflevector(lo, hi, 0,1,2,3,4,5,6,7,8,9,10,11,12,13,14,15);
}

// bare v_exp_f32 (no libm fixup); args here are always <= 0, underflow->0 is fine
__device__ __forceinline__ float fast_exp2(float x) {
  return __builtin_amdgcn_exp2f(x);
}

__global__ __launch_bounds__(NTHREADS)
void fa_fwd_f16wmma_t(const float* __restrict__ q,
                      const float* __restrict__ kv,
                      float* __restrict__ out)
{
  __shared__ _Float16 sK [BN * ROWH];            // K tile, row-major (key x d)
  __shared__ _Float16 sVt[Dc * ROWH];            // V tile, transposed (d x key)
  __shared__ _Float16 sP [NWAVES * 16 * ROWH];   // per-wave P^T scratch (query x key)

  const int tid  = threadIdx.x;
  const int wave = tid >> 5;
  const int lane = tid & 31;
  const int half = lane >> 4;   // selects K/M sub-ranges inside WMMA fragments
  const int l16  = lane & 15;   // this lane's QUERY (column) in the transposed tiles

  const int qt0 = blockIdx.x * BM;
  const int h   = blockIdx.y;
  const int b   = blockIdx.z;

  // softmax is base-invariant: fold 1/sqrt(64) AND log2(e) into Q, use exp2
  // everywhere -> every exponential is a single v_exp_f32.
  const float scale = 0.125f * 1.44269504088896340736f;

  const float* qbase  = q   + (size_t)b * Tc * QSTRIDE  + (size_t)h * Dc;
  const float* kvbase = kv  + (size_t)b * Tc * KVSTRIDE + (size_t)h * Dc;
  float*       obase  = out + (size_t)b * Tc * QSTRIDE  + (size_t)h * Dc;

  // ---- Q as B-fragments (32x16: head-dim x query). Lane n = query l16 holds
  //      head-dims kk*32 + half*16 + 0..15, i.e. 16 contiguous floats. ----
  v16h bQ[2];
  {
    const float* qrow = qbase + (size_t)(qt0 + wave * 16 + l16) * QSTRIDE;
    #pragma unroll
    for (int kk = 0; kk < 2; ++kk) {
      const float* p0 = qrow + kk * 32 + half * 16;
      #pragma unroll
      for (int a = 0; a < 4; ++a) {
        v4f f = *(const v4f*)(p0 + 4 * a);
        #pragma unroll
        for (int i = 0; i < 4; ++i) bQ[kk][4 * a + i] = (_Float16)(f[i] * scale);
      }
    }
  }

  // per-lane softmax state (log2 domain): this lane's query.
  // acc = O^T C-fragments (4 d-blocks).
  float m_run = -1e30f, l_run = 0.0f;
  v8f acc[4];
  #pragma unroll
  for (int d = 0; d < 4; ++d)
    #pragma unroll
    for (int i = 0; i < 8; ++i) acc[d][i] = 0.0f;

  constexpr int NT = Tc / BN;  // 32 kv tiles
  for (int nt = 0; nt < NT; ++nt) {
    const float* kp = kvbase + (size_t)(nt * BN) * KVSTRIDE;  // k plane
    const float* vp = kp + QSTRIDE;                           // v plane (+H*D)

    __syncthreads();  // previous-iteration readers finished with sK/sVt

    // ---- K tile: f32 -> f16, row-major, packed b64 stores ----
    #pragma unroll
    for (int it = 0; it < 8; ++it) {
      int idx = tid + it * NTHREADS;   // float4 index 0..1023
      int row = idx >> 4;
      int c4  = (idx & 15) << 2;
      v4f kf = *(const v4f*)(kp + (size_t)row * KVSTRIDE + c4);
      v4h kh;
      #pragma unroll
      for (int j = 0; j < 4; ++j) kh[j] = (_Float16)kf[j];
      *(v4h*)(&sK[row * ROWH + c4]) = kh;
    }
    // ---- V tile: transpose via key-pairs so stores are packed b32 ----
    #pragma unroll
    for (int it = 0; it < 4; ++it) {
      int p  = tid + it * NTHREADS;    // pair index 0..511
      int dq = p & 15;                 // d quad (x4)
      int kpair = p >> 4;              // key pair 0..31
      const float* v0 = vp + (size_t)(2 * kpair) * KVSTRIDE + 4 * dq;
      v4f f0 = *(const v4f*)(v0);
      v4f f1 = *(const v4f*)(v0 + KVSTRIDE);
      #pragma unroll
      for (int j = 0; j < 4; ++j) {
        v2h pk; pk[0] = (_Float16)f0[j]; pk[1] = (_Float16)f1[j];
        *(v2h*)(&sVt[(4 * dq + j) * ROWH + 2 * kpair]) = pk;
      }
    }
    if (nt + 1 < NT) {  // uniform: prefetch next tile (global_prefetch_b8)
      const float* kn = kp + (size_t)BN * KVSTRIDE;
      __builtin_prefetch(kn + (size_t)(tid >> 1) * KVSTRIDE + (size_t)(tid & 1) * 32, 0, 1);
      __builtin_prefetch(kn + QSTRIDE + (size_t)(tid >> 1) * KVSTRIDE + (size_t)(tid & 1) * 32, 0, 1);
    }
    __syncthreads();

    // ---- S^T = K * Q^T : 4 key-blocks of 16x16; lane column = query ----
    v8f st[4];
    #pragma unroll
    for (int nb = 0; nb < 4; ++nb) {
      const _Float16* kr = &sK[(nb * 16 + l16) * ROWH + half * 16];
      v16h aK0 = cat8(*(const v8h*)(kr),      *(const v8h*)(kr + 8));   // dims 0..31
      v16h aK1 = cat8(*(const v8h*)(kr + 32), *(const v8h*)(kr + 40));  // dims 32..63
      v8f c;
      #pragma unroll
      for (int i = 0; i < 8; ++i) c[i] = 0.0f;
      c = __builtin_amdgcn_wmma_f32_16x16x32_f16(false, aK0, false, bQ[0], (short)0, c, false, false);
      c = __builtin_amdgcn_wmma_f32_16x16x32_f16(false, aK1, false, bQ[1], (short)0, c, false, false);
      st[nb] = c;
    }

    // ---- online softmax (base 2): all stats per-lane (this lane's query) ----
    float mx = st[0][0];
    #pragma unroll
    for (int nb = 0; nb < 4; ++nb)
      #pragma unroll
      for (int i = 0; i < 8; ++i) mx = fmaxf(mx, st[nb][i]);
    mx = fmaxf(mx, __shfl_xor(mx, 16, 32));       // merge the two key-halves

    const float mn   = fmaxf(m_run, mx);
    const float corr = fast_exp2(m_run - mn);
    m_run = mn;
    l_run *= corr;
    #pragma unroll
    for (int d = 0; d < 4; ++d)
      #pragma unroll
      for (int i = 0; i < 8; ++i) acc[d][i] *= corr;

    // P^T = exp2(S^T - m): in-lane sum, stage rows of P^T into per-wave LDS
    _Float16* pw = &sP[(wave * 16 + l16) * ROWH];
    float rs = 0.0f;
    #pragma unroll
    for (int nb = 0; nb < 4; ++nb) {
      v8h ph;
      #pragma unroll
      for (int i = 0; i < 8; ++i) {
        float pv = fast_exp2(st[nb][i] - mn);
        rs += pv;
        ph[i] = (_Float16)pv;
      }
      *(v8h*)(pw + nb * 16 + half * 8) = ph;      // keys nb*16+half*8 .. +7
    }
    rs += __shfl_xor(rs, 16, 32);                 // merge key-halves
    l_run += rs;

    // wave-local fence: our DS stores of P^T retire before re-reading as B-fragments
    asm volatile("s_wait_dscnt 0" ::: "memory");

    // P^T as B-fragments (keys x query): lane = query, keys contiguous per chunk
    const _Float16* pr = &sP[(wave * 16 + l16) * ROWH];
    v16h bP0 = cat8(*(const v8h*)(pr + half * 16),      *(const v8h*)(pr + half * 16 + 8));
    v16h bP1 = cat8(*(const v8h*)(pr + 32 + half * 16), *(const v8h*)(pr + 32 + half * 16 + 8));

    // ---- O^T += V^T * P^T : A-fragments from sVt rows (contiguous) ----
    #pragma unroll
    for (int db = 0; db < 4; ++db) {
      const _Float16* vr = &sVt[(db * 16 + l16) * ROWH + half * 16];
      v16h aV0 = cat8(*(const v8h*)(vr),      *(const v8h*)(vr + 8));   // keys 0..31
      v16h aV1 = cat8(*(const v8h*)(vr + 32), *(const v8h*)(vr + 40));  // keys 32..63
      acc[db] = __builtin_amdgcn_wmma_f32_16x16x32_f16(false, aV0, false, bP0, (short)0, acc[db], false, false);
      acc[db] = __builtin_amdgcn_wmma_f32_16x16x32_f16(false, aV1, false, bP1, (short)0, acc[db], false, false);
    }
  }

  // ---- epilogue: per-lane normalize (lane = query), fully vectorized stores ----
  const float inv = 1.0f / l_run;
  float* o = obase + (size_t)(qt0 + wave * 16 + l16) * QSTRIDE + half * 8;
  #pragma unroll
  for (int db = 0; db < 4; ++db) {
    v4f o0, o1;
    #pragma unroll
    for (int i = 0; i < 4; ++i) { o0[i] = acc[db][i] * inv; o1[i] = acc[db][4 + i] * inv; }
    *(v4f*)(o + db * 16)     = o0;   // d = db*16 + half*8 + 0..3
    *(v4f*)(o + db * 16 + 4) = o1;   // d = db*16 + half*8 + 4..7
  }
}

extern "C" void kernel_launch(void* const* d_in, const int* in_sizes, int n_in,
                              void* d_out, int out_size, void* d_ws, size_t ws_size,
                              hipStream_t stream) {
  (void)in_sizes; (void)n_in; (void)out_size; (void)d_ws; (void)ws_size;
  const float* q  = (const float*)d_in[0];
  const float* kv = (const float*)d_in[1];
  float* out = (float*)d_out;
  dim3 grid(Tc / BM, Hc, Bc);   // (32, 16, 2) workgroups
  dim3 block(NTHREADS);         // 128 threads = 4 wave32
  hipLaunchKernelGGL(fa_fwd_f16wmma_t, grid, block, 0, stream, q, kv, out);
}
